// InteractionFFN_19035295056635
// MI455X (gfx1250) — matrix operand order repaired and checked by reading, not compiled, not run on hardware
//
#include <hip/hip_runtime.h>
#include <hip/hip_bf16.h>
#include <math.h>

// ---------------------------------------------------------------------------
// InteractionFFN for MI455X (gfx1250, wave32, WMMA).
// Heavy GEMMs: v_wmma_f32_16x16x32_bf16, 32x64 register-blocked per wave
// (2 M-tiles x 4 N-tiles = 8 accumulators) to cut L2 traffic 4x on A / 2x on B.
// MoE low-rank path densified into two dense GEMMs.
// ---------------------------------------------------------------------------

typedef __attribute__((ext_vector_type(16))) __bf16 bf16x16;
typedef __attribute__((ext_vector_type(8)))  __bf16 bf16x8;
typedef __attribute__((ext_vector_type(8)))  float  f32x8;

static __device__ __host__ inline __bf16 f2bf(float f) {
  unsigned u = __builtin_bit_cast(unsigned, f);
  unsigned r = (u + 0x7FFFu + ((u >> 16) & 1u)) >> 16;   // round-to-nearest-even
  return __builtin_bit_cast(__bf16, (unsigned short)r);
}
static __device__ __host__ inline float bf2f(__bf16 b) {
  unsigned short s = __builtin_bit_cast(unsigned short, b);
  return __builtin_bit_cast(float, (unsigned)s << 16);
}

static __device__ inline bf16x16 load_a_frag(const __bf16* p) {
  // A fragment: lane holds row M, elements {k+8g..+7} then {k+16+8g..+7}
  bf16x8 lo = *(const bf16x8*)(p);
  bf16x8 hi = *(const bf16x8*)(p + 16);
  return __builtin_shufflevector(lo, hi, 0,1,2,3,4,5,6,7,8,9,10,11,12,13,14,15);
}
static __device__ inline bf16x16 load_b_frag(const __bf16* p) {
  // B fragment: lane holds column N (= row of Bt), elements {k+8g..+7} then {k+16+8g..+7}
  bf16x8 lo = *(const bf16x8*)(p);
  bf16x8 hi = *(const bf16x8*)(p + 16);
  return __builtin_shufflevector(lo, hi, 0,1,2,3,4,5,6,7,8,9,10,11,12,13,14,15);
}

// ---------------------------------------------------------------------------
// bf16 WMMA GEMM:  C[M,N] = A[M,K] * Bt[N,K]^T  (+bias, +sigmoid)
// 8 waves (256 threads) per block; each wave owns a 32x64 output macro-tile.
// EPI: 0 = none, 1 = +bias, 2 = sigmoid(x + bias)
// Requires M % 32 == 0, N % 64 == 0, K % 32 == 0.
// ---------------------------------------------------------------------------
template <int EPI, bool OUT_BF16>
__global__ __launch_bounds__(256) void gemm_wmma_bf16(
    const __bf16* __restrict__ A, const __bf16* __restrict__ Bt,
    const float* __restrict__ bias, void* __restrict__ Cout,
    int M, int N, int Kd)
{
  const int lane = threadIdx.x & 31;
  const int wave = threadIdx.x >> 5;
  const int tilesN4 = N >> 6;                  // macro-tiles along N
  const int tile = blockIdx.x * 8 + wave;
  const int tm = tile / tilesN4;               // 32-row macro tile index
  const int tn = tile - tm * tilesN4;          // 64-col macro tile index
  const int r  = lane & 15;
  const int g  = lane >> 4;

  const __bf16* ap0 = A  + (size_t)(tm * 32 + r)      * Kd + g * 8;
  const __bf16* ap1 = A  + (size_t)(tm * 32 + 16 + r) * Kd + g * 8;
  const __bf16* bp  = Bt + (size_t)(tn * 64 + r)      * Kd + g * 8;
  const size_t bstride = (size_t)16 * Kd;      // rows between adjacent N-tiles

  f32x8 acc[2][4] = {};
  for (int kb = 0; kb < Kd; kb += 32) {
    // near-cache (WGP-scope) prefetch of the next A chunks -> global_prefetch_b8
    __builtin_prefetch(ap0 + 128, 0, 3);
    __builtin_prefetch(ap1 + 128, 0, 3);
    bf16x16 a0 = load_a_frag(ap0);
    bf16x16 a1 = load_a_frag(ap1);
#pragma unroll
    for (int n = 0; n < 4; ++n) {
      bf16x16 b = load_b_frag(bp + n * bstride);
      acc[0][n] = __builtin_amdgcn_wmma_f32_16x16x32_bf16(
          false, a0, false, b, (short)0, acc[0][n], false, false);
      acc[1][n] = __builtin_amdgcn_wmma_f32_16x16x32_bf16(
          false, a1, false, b, (short)0, acc[1][n], false, false);
    }
    ap0 += 32; ap1 += 32; bp += 32;
  }

#pragma unroll
  for (int mi = 0; mi < 2; ++mi) {
    const size_t row0 = (size_t)(tm * 32 + mi * 16 + g * 8);
#pragma unroll
    for (int n = 0; n < 4; ++n) {
      const int col = tn * 64 + n * 16 + r;
      float bv = (EPI >= 1) ? bias[col] : 0.0f;
#pragma unroll
      for (int v = 0; v < 8; ++v) {
        float val = acc[mi][n][v] + bv;
        if (EPI == 2) val = 1.0f / (1.0f + expf(-val));
        if (OUT_BF16) ((__bf16*)Cout)[(row0 + v) * N + col] = f2bf(val);
        else          ((float*)Cout)[(row0 + v) * N + col] = val;
      }
    }
  }
}

// ---------------------------------------------------------------------------
// Conversion / layout kernels (one-time, small)
// ---------------------------------------------------------------------------
__global__ void cast_bf16_kernel(const float* __restrict__ s,
                                 __bf16* __restrict__ d, int n) {
  int idx = blockIdx.x * blockDim.x + threadIdx.x;
  if (idx < n) d[idx] = f2bf(s[idx]);
}

// dst[c*rows + r] = bf16(src[r*cols + c])   (src: [rows,cols] f32)
__global__ void transpose_bf16_kernel(const float* __restrict__ s,
                                      __bf16* __restrict__ d,
                                      int rows, int cols, int n) {
  int idx = blockIdx.x * blockDim.x + threadIdx.x;
  if (idx >= n) return;
  int c = idx / rows;
  int r = idx - c * rows;
  d[idx] = f2bf(s[(size_t)r * cols + c]);
}

// A_flat_t[(p*32+r)*512 + d] = bf16(pattern_up_A[p][d][r])   ([P,D,R] f32)
__global__ void aflat_kernel(const float* __restrict__ A,
                             __bf16* __restrict__ d, int n) {
  int idx = blockIdx.x * blockDim.x + threadIdx.x;
  if (idx >= n) return;
  int dc = idx & 511;
  int pr = idx >> 9;
  int p  = pr >> 5;
  int rr = pr & 31;
  d[idx] = f2bf(A[((size_t)p * 512 + dc) * 32 + rr]);
}

// ---------------------------------------------------------------------------
// Per-token cross-neuron gating attention + aggregation + pattern routing.
// One block (256 threads) per token; q/k/v tiles resident in LDS.
// Produces combined (bf16) and dense pattern weights wdense[T,32]
// (softmax(top4) scattered, zeros elsewhere).
// ---------------------------------------------------------------------------
__global__ __launch_bounds__(256) void attn_route_kernel(
    const __bf16* __restrict__ Qb, const __bf16* __restrict__ Kb,
    const __bf16* __restrict__ Vb,
    const float* __restrict__ sn,    // selected_neurons [T,8,512]
    const float* __restrict__ topw,  // topk_neuron_weights [T,8]
    const float* __restrict__ ctx,   // context [T,512]
    const float* __restrict__ xin,   // x [T,512]
    const float* __restrict__ pq,    // pattern_queries [32,512]
    __bf16* __restrict__ comb_b,     // combined bf16 [T,512]
    float* __restrict__ wdense)      // dense pattern weights [T,32]
{
  constexpr int D = 512, Kn = 8, P = 32;
  const int t = blockIdx.x;
  const int tid = threadIdx.x;

  __shared__ float qs[Kn * D];
  __shared__ float ks[Kn * D];
  __shared__ float vs[Kn * D];     // already sigmoided by GEMM epilogue
  __shared__ float sc[512];        // scores/attn [h][i][j]
  __shared__ float agg[D];
  __shared__ float ps[P];

  const size_t base = (size_t)t * Kn * D;
  for (int i = tid; i < Kn * D; i += 256) {
    qs[i] = bf2f(Qb[base + i]);
    ks[i] = bf2f(Kb[base + i]);
    vs[i] = bf2f(Vb[base + i]);
  }
  if (tid < P) wdense[t * P + tid] = 0.0f;
  __syncthreads();

  // scores[h,i,j] = (q_i,h . k_j,h) / sqrt(64)
  for (int z = tid; z < 512; z += 256) {
    int h = z >> 6, i = (z >> 3) & 7, j = z & 7;
    const float* qrow = qs + i * D + h * 64;
    const float* krow = ks + j * D + h * 64;
    float s = 0.f;
#pragma unroll 8
    for (int dd = 0; dd < 64; ++dd) s += qrow[dd] * krow[dd];
    sc[z] = s * 0.125f;
  }
  __syncthreads();

  // softmax over j (8-wide rows)
  if (tid < 64) {
    float* row = sc + tid * 8;
    float m = row[0];
#pragma unroll
    for (int j = 1; j < 8; ++j) m = fmaxf(m, row[j]);
    float ssum = 0.f;
#pragma unroll
    for (int j = 0; j < 8; ++j) { float e = expf(row[j] - m); row[j] = e; ssum += e; }
    float inv = 1.0f / ssum;
#pragma unroll
    for (int j = 0; j < 8; ++j) row[j] *= inv;
  }
  __syncthreads();

  // gates, modulation, weighted aggregation, combined = x + aggregated
  float wk[8];
#pragma unroll
  for (int i = 0; i < 8; ++i) wk[i] = topw[t * 8 + i];
  for (int d = tid; d < D; d += 256) {
    int h = d >> 6;
    float acc = 0.f;
#pragma unroll
    for (int i = 0; i < 8; ++i) {
      const float* arow = sc + (h << 6) + (i << 3);
      float gi = 0.f;
#pragma unroll
      for (int j = 0; j < 8; ++j) gi += arow[j] * vs[j * D + d];
      acc += wk[i] * sn[base + (size_t)i * D + d] * gi;
    }
    agg[d] = acc;
    float cb = xin[(size_t)t * D + d] + acc;
    comb_b[(size_t)t * D + d] = f2bf(cb);
  }
  __syncthreads();

  // pattern scores: 0.5 * (agg . pq_p)/sqrt(D) + 0.5 * (ctx . pq_p)
  if (tid < P) {
    int p = tid;
    float ns = 0.f, cs = 0.f;
    for (int d = 0; d < D; ++d) {
      float pv = pq[p * D + d];
      ns += agg[d] * pv;
      cs += ctx[(size_t)t * D + d] * pv;
    }
    ps[p] = 0.5f * ns * 0.044194173824159216f + 0.5f * cs;  // 1/sqrt(512)
  }
  __syncthreads();

  // top-4 + softmax, scattered to dense weights
  if (tid == 0) {
    int id[4]; float sv[4];
    unsigned taken = 0;
    for (int kk = 0; kk < 4; ++kk) {
      float best = -3.4e38f; int bi = 0;
      for (int p = 0; p < P; ++p)
        if (!((taken >> p) & 1u) && ps[p] > best) { best = ps[p]; bi = p; }
      taken |= 1u << bi; id[kk] = bi; sv[kk] = best;
    }
    float m = sv[0], ssum = 0.f, w[4];
#pragma unroll
    for (int kk = 0; kk < 4; ++kk) { w[kk] = expf(sv[kk] - m); ssum += w[kk]; }
    float inv = 1.0f / ssum;
#pragma unroll
    for (int kk = 0; kk < 4; ++kk) wdense[t * P + id[kk]] = w[kk] * inv;
  }
}

// g[t, p*32+r] = bf16( wdense[t,p] * h_mid_all[t, p*32+r] )
__global__ void scale_g_kernel(const float* __restrict__ hmid,
                               const float* __restrict__ wdense,
                               __bf16* __restrict__ g, int n) {
  int idx = blockIdx.x * blockDim.x + threadIdx.x;
  if (idx >= n) return;
  int t = idx >> 10;
  int p = (idx >> 5) & 31;
  g[idx] = f2bf(wdense[t * 32 + p] * hmid[idx]);
}

// h = gelu_exact(0.1*h_base + 0.9*h_pattern)  -> bf16 for the down GEMM
__global__ void act_kernel(const float* __restrict__ hb,
                           const float* __restrict__ hp,
                           __bf16* __restrict__ out, int n) {
  int idx = blockIdx.x * blockDim.x + threadIdx.x;
  if (idx >= n) return;
  float h = 0.1f * hb[idx] + 0.9f * hp[idx];
  float ge = 0.5f * h * (1.0f + erff(h * 0.70710678118654752f));
  out[idx] = f2bf(ge);
}

// ---------------------------------------------------------------------------
// Host driver
// ---------------------------------------------------------------------------
extern "C" void kernel_launch(void* const* d_in, const int* in_sizes, int n_in,
                              void* d_out, int out_size, void* d_ws, size_t ws_size,
                              hipStream_t stream) {
  (void)in_sizes; (void)n_in; (void)out_size; (void)ws_size;
  constexpr int T  = 2048;          // B*S
  constexpr int D  = 512;
  constexpr int F  = 1024;
  constexpr int Kn = 8;
  constexpr int TK = T * Kn;        // 16384 projection rows
  constexpr int P  = 32;
  constexpr int PR = 1024;          // P*R

  const float* x    = (const float*)d_in[0];
  const float* sn   = (const float*)d_in[1];
  const float* topw = (const float*)d_in[2];
  const float* ctx  = (const float*)d_in[3];
  const float* Wq   = (const float*)d_in[4];
  const float* bq   = (const float*)d_in[5];
  const float* Wk   = (const float*)d_in[6];
  const float* bk   = (const float*)d_in[7];
  const float* Wv   = (const float*)d_in[8];
  const float* bv   = (const float*)d_in[9];
  const float* pq   = (const float*)d_in[10];
  const float* pA   = (const float*)d_in[11];
  const float* pB   = (const float*)d_in[12];
  const float* Wup  = (const float*)d_in[13];
  const float* bup  = (const float*)d_in[14];
  const float* Wdn  = (const float*)d_in[15];
  const float* bdn  = (const float*)d_in[16];
  float* out = (float*)d_out;

  char* ws = (char*)d_ws;
  size_t off = 0;
  auto alloc = [&](size_t bytes) -> void* {
    off = (off + 255) & ~(size_t)255;
    void* p = ws + off;
    off += bytes;
    return p;
  };

  __bf16* Xbf  = (__bf16*)alloc((size_t)TK * D * 2);
  __bf16* Wqt  = (__bf16*)alloc((size_t)D * D * 2);
  __bf16* Wkt  = (__bf16*)alloc((size_t)D * D * 2);
  __bf16* Wvt  = (__bf16*)alloc((size_t)D * D * 2);
  __bf16* Wupt = (__bf16*)alloc((size_t)F * D * 2);
  __bf16* Wdnt = (__bf16*)alloc((size_t)D * F * 2);
  __bf16* Aft  = (__bf16*)alloc((size_t)PR * D * 2);
  __bf16* Bft  = (__bf16*)alloc((size_t)F * PR * 2);
  __bf16* Qb   = (__bf16*)alloc((size_t)TK * D * 2);
  __bf16* Kb   = (__bf16*)alloc((size_t)TK * D * 2);
  __bf16* Vb   = (__bf16*)alloc((size_t)TK * D * 2);
  __bf16* Cb   = (__bf16*)alloc((size_t)T * D * 2);    // combined bf16
  float*  Wd   = (float*) alloc((size_t)T * P * 4);    // dense pattern weights
  float*  Hmid = (float*) alloc((size_t)T * PR * 4);
  __bf16* Gbf  = (__bf16*)alloc((size_t)T * PR * 2);
  float*  Hbas = (float*) alloc((size_t)T * F * 4);
  float*  Hpat = (float*) alloc((size_t)T * F * 4);
  __bf16* Hbf  = (__bf16*)alloc((size_t)T * F * 2);

  const int TPB = 256;
  auto blocks = [](int n) { return (n + 255) / 256; };
  // macro-tiles: 32 rows x 64 cols per wave, 8 waves per block
  auto gemm_blocks = [](int M, int N) { return (M / 32) * (N / 64) / 8; };

  // --- precision conversions / weight transposes (one-time, small) ---
  cast_bf16_kernel<<<blocks(TK * D), TPB, 0, stream>>>(sn, Xbf, TK * D);
  transpose_bf16_kernel<<<blocks(D * D), TPB, 0, stream>>>(Wq, Wqt, D, D, D * D);
  transpose_bf16_kernel<<<blocks(D * D), TPB, 0, stream>>>(Wk, Wkt, D, D, D * D);
  transpose_bf16_kernel<<<blocks(D * D), TPB, 0, stream>>>(Wv, Wvt, D, D, D * D);
  transpose_bf16_kernel<<<blocks(D * F), TPB, 0, stream>>>(Wup, Wupt, D, F, D * F);
  transpose_bf16_kernel<<<blocks(F * D), TPB, 0, stream>>>(Wdn, Wdnt, F, D, F * D);
  transpose_bf16_kernel<<<blocks(PR * F), TPB, 0, stream>>>(pB, Bft, PR, F, PR * F);
  aflat_kernel<<<blocks(PR * D), TPB, 0, stream>>>(pA, Aft, PR * D);

  // --- Q/K/V projections (dominant GEMMs): [16384,512]@[512,512] ---
  gemm_wmma_bf16<1, true><<<gemm_blocks(TK, D), TPB, 0, stream>>>(Xbf, Wqt, bq, Qb, TK, D, D);
  gemm_wmma_bf16<1, true><<<gemm_blocks(TK, D), TPB, 0, stream>>>(Xbf, Wkt, bk, Kb, TK, D, D);
  gemm_wmma_bf16<2, true><<<gemm_blocks(TK, D), TPB, 0, stream>>>(Xbf, Wvt, bv, Vb, TK, D, D);

  // --- gating attention + aggregation + pattern routing (per token) ---
  attn_route_kernel<<<T, TPB, 0, stream>>>(Qb, Kb, Vb, sn, topw, ctx, x, pq, Cb, Wd);

  // --- base up projection: [2048,512]@[512,1024] ---
  gemm_wmma_bf16<1, false><<<gemm_blocks(T, F), TPB, 0, stream>>>(
      Cb, Wupt, bup, Hbas, T, F, D);

  // --- densified expert path: mid = combined @ A_flat  [2048,512]@[512,1024] ---
  gemm_wmma_bf16<0, false><<<gemm_blocks(T, PR), TPB, 0, stream>>>(
      Cb, Aft, nullptr, Hmid, T, PR, D);

  // g = wdense * mid (zeros kill non-selected patterns)
  scale_g_kernel<<<blocks(T * PR), TPB, 0, stream>>>(Hmid, Wd, Gbf, T * PR);

  // h_pattern = g @ B_flat  [2048,1024]@[1024,1024]
  gemm_wmma_bf16<0, false><<<gemm_blocks(T, F), TPB, 0, stream>>>(
      Gbf, Bft, nullptr, Hpat, T, F, PR);

  // h = gelu(0.1*h_base + 0.9*h_pattern)
  act_kernel<<<blocks(T * F), TPB, 0, stream>>>(Hbas, Hpat, Hbf, T * F);

  // --- down projection: [2048,1024]@[1024,512] + b_down -> f32 output ---
  gemm_wmma_bf16<1, false><<<gemm_blocks(T, D), TPB, 0, stream>>>(
      Hbf, Wdnt, bdn, out, T, D, F);
}
